// RNN_11527692223012
// MI455X (gfx1250) — compile-verified, matrix operand order checked by reading, not run on hardware
//
#include <hip/hip_runtime.h>
#include <math.h>

// ---------------- CDNA5 WMMA types ----------------
typedef __bf16 v16bf __attribute__((ext_vector_type(16)));
typedef float  v8f   __attribute__((ext_vector_type(8)));
typedef unsigned int v4u __attribute__((ext_vector_type(4)));

constexpr int BB   = 64;    // batch
constexpr int TT   = 1000;  // timesteps
constexpr int KIN0 = 256;   // layer0 input features
constexpr int HH   = 512;   // hidden
constexpr int GG   = 1536;  // 3*HH gate rows
constexpr int NWG  = 32;    // 512/16 column tiles
constexpr int THREADS = 128;

// ---------------- bf16 split helpers ----------------
__device__ __forceinline__ unsigned short f2bf(float f) {
  unsigned u = __float_as_uint(f);
  unsigned r = u + 0x7FFFu + ((u >> 16) & 1u);   // round-to-nearest-even
  return (unsigned short)(r >> 16);
}
__device__ __forceinline__ float bf2f(unsigned short h) {
  return __uint_as_float(((unsigned)h) << 16);
}

// Fragment: 16 bf16 per lane (A 16x32 or B 32x16 operand)
union Frag {
  v16bf v;
  v4u   q[2];
};

// Load a 16-bf16 fragment half-pair from a row-major bf16 plane.
// Per ISA layout: lane kbase = k0 + 8*(lane>>4); chunk0 = [kbase..+7], chunk1 = [kbase+16..+23]
__device__ __forceinline__ void load2(const unsigned short* __restrict__ row, int kb, Frag& f) {
  f.q[0] = *(const v4u*)(row + kb);
  f.q[1] = *(const v4u*)(row + kb + 16);
}

// fp32-accurate product via 3 bf16 WMMAs: C += Ah*Bl + Al*Bh + Ah*Bh
__device__ __forceinline__ v8f wmma3(const Frag& ah, const Frag& al,
                                     const Frag& bh, const Frag& bl, v8f c) {
  c = __builtin_amdgcn_wmma_f32_16x16x32_bf16(false, ah.v, false, bl.v, (short)0, c, false, false);
  c = __builtin_amdgcn_wmma_f32_16x16x32_bf16(false, al.v, false, bh.v, (short)0, c, false, false);
  c = __builtin_amdgcn_wmma_f32_16x16x32_bf16(false, ah.v, false, bh.v, (short)0, c, false, false);
  return c;
}

__device__ __forceinline__ float sigmoidf_(float x) {
  return 1.0f / (1.0f + __expf(-x));
}

// ---------------- small utility kernels ----------------
__global__ void convert_planes(const float* __restrict__ src,
                               unsigned short* __restrict__ hi,
                               unsigned short* __restrict__ lo, long long n) {
  long long i = (long long)blockIdx.x * blockDim.x + threadIdx.x;
  if (i < n) {
    float f = src[i];
    unsigned short h = f2bf(f);
    hi[i] = h;
    lo[i] = f2bf(f - bf2f(h));
  }
}

__global__ void zero_u32(unsigned* __restrict__ p, long long n) {
  long long i = (long long)blockIdx.x * blockDim.x + threadIdx.x;
  if (i < n) p[i] = 0u;
}

// ---------------- persistent GRU layer kernel ----------------
// grid = NWG (32) workgroups, 128 threads (4 wave32). WG owns hidden columns
// [16*blockIdx.x, +16). Wave w owns batch rows [16*w, +16).
template <int KIN, bool WRITE_SEQ>
__global__ __launch_bounds__(THREADS, 1)
void gru_kernel(const unsigned short* __restrict__ ain_hi,   // input sequence bf16 planes
                const unsigned short* __restrict__ ain_lo,
                long long t_stride, long long b_stride,       // element strides into ain
                const unsigned short* __restrict__ wih_hi,    // [GG*KIN] bf16 planes
                const unsigned short* __restrict__ wih_lo,
                const unsigned short* __restrict__ whh_hi,    // [GG*HH]
                const unsigned short* __restrict__ whh_lo,
                const float* __restrict__ b_ih,
                const float* __restrict__ b_hh,
                unsigned short* __restrict__ hb_hi,           // [2*BB*HH] ping-pong h
                unsigned short* __restrict__ hb_lo,
                unsigned short* __restrict__ seq_hi,          // layer0: [TT*BB*HH] out planes
                unsigned short* __restrict__ seq_lo,
                float* __restrict__ osum,                     // layer1: [BB*HH]
                float* __restrict__ olast,                    // layer1: [BB*HH]
                unsigned* __restrict__ bar)                   // [0]=count, [1]=generation
{
  const int tid  = threadIdx.x;
  const int wave = tid >> 5;
  const int lane = tid & 31;
  const int lh   = lane >> 4;    // lane half
  const int ln   = lane & 15;
  const int j    = blockIdx.x * 16 + ln;   // hidden column owned by this lane (B/C "N")
  const int arow = 16 * wave + ln;         // batch row this lane loads for A

  // weight row bases for the three gates (r, z, n)
  const size_t rowR_i = (size_t)(j)        * KIN;
  const size_t rowZ_i = (size_t)(HH + j)   * KIN;
  const size_t rowN_i = (size_t)(2*HH + j) * KIN;
  const size_t rowR_h = (size_t)(j)        * HH;
  const size_t rowZ_h = (size_t)(HH + j)   * HH;
  const size_t rowN_h = (size_t)(2*HH + j) * HH;

  const float bias_r  = b_ih[j]        + b_hh[j];
  const float bias_z  = b_ih[HH + j]   + b_hh[HH + j];
  const float bias_xn = b_ih[2*HH + j];
  const float bias_hn = b_hh[2*HH + j];

  v8f oacc = {0.f,0.f,0.f,0.f,0.f,0.f,0.f,0.f};

  for (int t = 0; t < TT; ++t) {
    const int cur = t & 1;
    const unsigned short* hprev_hi = hb_hi + (size_t)cur * (BB * HH);
    const unsigned short* hprev_lo = hb_lo + (size_t)cur * (BB * HH);
    unsigned short* hnext_hi = hb_hi + (size_t)(cur ^ 1) * (BB * HH);
    unsigned short* hnext_lo = hb_lo + (size_t)(cur ^ 1) * (BB * HH);

    v8f accR  = {0.f,0.f,0.f,0.f,0.f,0.f,0.f,0.f};
    v8f accZ  = {0.f,0.f,0.f,0.f,0.f,0.f,0.f,0.f};
    v8f accXN = {0.f,0.f,0.f,0.f,0.f,0.f,0.f,0.f};
    v8f accHN = {0.f,0.f,0.f,0.f,0.f,0.f,0.f,0.f};

    // ---- input projection: x_t @ W_ih^T (K = KIN) ----
    {
      const unsigned short* a_hi = ain_hi + (size_t)t * t_stride + (size_t)arow * b_stride;
      const unsigned short* a_lo = ain_lo + (size_t)t * t_stride + (size_t)arow * b_stride;
      for (int k0 = 0; k0 < KIN; k0 += 32) {
        const int kb = k0 + 8 * lh;
        Frag ah, al, bh, bl;
        load2(a_hi, kb, ah); load2(a_lo, kb, al);
        load2(wih_hi + rowR_i, kb, bh); load2(wih_lo + rowR_i, kb, bl);
        accR = wmma3(ah, al, bh, bl, accR);
        load2(wih_hi + rowZ_i, kb, bh); load2(wih_lo + rowZ_i, kb, bl);
        accZ = wmma3(ah, al, bh, bl, accZ);
        load2(wih_hi + rowN_i, kb, bh); load2(wih_lo + rowN_i, kb, bl);
        accXN = wmma3(ah, al, bh, bl, accXN);
      }
    }

    // ---- recurrent projection: h_{t-1} @ W_hh^T (K = HH) ----
    {
      const unsigned short* a_hi = hprev_hi + (size_t)arow * HH;
      const unsigned short* a_lo = hprev_lo + (size_t)arow * HH;
      for (int k0 = 0; k0 < HH; k0 += 32) {
        const int kb = k0 + 8 * lh;
        Frag ah, al, bh, bl;
        load2(a_hi, kb, ah); load2(a_lo, kb, al);
        load2(whh_hi + rowR_h, kb, bh); load2(whh_lo + rowR_h, kb, bl);
        accR = wmma3(ah, al, bh, bl, accR);
        load2(whh_hi + rowZ_h, kb, bh); load2(whh_lo + rowZ_h, kb, bl);
        accZ = wmma3(ah, al, bh, bl, accZ);
        load2(whh_hi + rowN_h, kb, bh); load2(whh_lo + rowN_h, kb, bl);
        accHN = wmma3(ah, al, bh, bl, accHN);
      }
    }

    // ---- gate math: C layout => element i is (b = 16*wave + i + 8*lh, col j) ----
#pragma unroll
    for (int i = 0; i < 8; ++i) {
      const int b = 16 * wave + i + 8 * lh;
      const size_t idx = (size_t)b * HH + j;
      const float hp = bf2f(hprev_hi[idx]) + bf2f(hprev_lo[idx]);
      const float r  = sigmoidf_(accR[i] + bias_r);
      const float z  = sigmoidf_(accZ[i] + bias_z);
      const float ng = tanhf(accXN[i] + bias_xn + r * (accHN[i] + bias_hn));
      const float h  = (1.0f - z) * ng + z * hp;
      const unsigned short hh = f2bf(h);
      const unsigned short hl = f2bf(h - bf2f(hh));
      hnext_hi[idx] = hh;
      hnext_lo[idx] = hl;
      if (WRITE_SEQ) {
        const size_t sidx = (size_t)t * (BB * HH) + idx;
        seq_hi[sidx] = hh;
        seq_lo[sidx] = hl;
      } else {
        oacc[i] += h;
        if (t == TT - 1) olast[idx] = h;
      }
    }

    // ---- grid-wide barrier between timesteps ----
    __threadfence();
    __syncthreads();
    if (tid == 0) {
      unsigned g = __hip_atomic_load(&bar[1], __ATOMIC_RELAXED, __HIP_MEMORY_SCOPE_AGENT);
      unsigned a = __hip_atomic_fetch_add(&bar[0], 1u, __ATOMIC_ACQ_REL, __HIP_MEMORY_SCOPE_AGENT);
      if (a == gridDim.x - 1) {
        __hip_atomic_store(&bar[0], 0u, __ATOMIC_RELAXED, __HIP_MEMORY_SCOPE_AGENT);
        __hip_atomic_fetch_add(&bar[1], 1u, __ATOMIC_RELEASE, __HIP_MEMORY_SCOPE_AGENT);
      } else {
        while (__hip_atomic_load(&bar[1], __ATOMIC_ACQUIRE, __HIP_MEMORY_SCOPE_AGENT) == g) {
          __builtin_amdgcn_s_sleep(2);
        }
      }
    }
    __syncthreads();
    __threadfence();
  }

  if (!WRITE_SEQ) {
#pragma unroll
    for (int i = 0; i < 8; ++i) {
      const int b = 16 * wave + i + 8 * lh;
      osum[(size_t)b * HH + j] = oacc[i];
    }
  }
}

// ---------------- SE block + FC tail (tiny) ----------------
__global__ __launch_bounds__(256)
void tail_kernel(const float* __restrict__ osum, const float* __restrict__ olast,
                 const float* __restrict__ se_w1, const float* __restrict__ se_b1,
                 const float* __restrict__ se_w2, const float* __restrict__ se_b2,
                 const float* __restrict__ fc1_w, const float* __restrict__ fc1_b,
                 const float* __restrict__ fc2_w, const float* __restrict__ fc2_b,
                 float* __restrict__ out)
{
  __shared__ float sm[HH];     // mean over time
  __shared__ float s1[HH / 8]; // SE squeeze
  __shared__ float slast[HH];  // gated last hidden
  __shared__ float shh[HH];    // fc1 output
  const int b = blockIdx.x;
  const int tid = threadIdx.x;

  for (int jj = tid; jj < HH; jj += 256) sm[jj] = osum[(size_t)b * HH + jj] * (1.0f / TT);
  __syncthreads();

  if (tid < HH / 8) {
    float acc = se_b1[tid];
    for (int k = 0; k < HH; ++k) acc += sm[k] * se_w1[(size_t)tid * HH + k];
    s1[tid] = fmaxf(acc, 0.0f);
  }
  __syncthreads();

  for (int jj = tid; jj < HH; jj += 256) {
    float acc = se_b2[jj];
    for (int q = 0; q < HH / 8; ++q) acc += s1[q] * se_w2[(size_t)jj * (HH / 8) + q];
    const float y = sigmoidf_(acc);
    const float o = olast[(size_t)b * HH + jj];
    slast[jj] = fmaxf(o * (1.0f + y), 0.0f);   // out*y + residual, relu
  }
  __syncthreads();

  for (int p = tid; p < HH; p += 256) {
    float acc = fc1_b[p];
    for (int k = 0; k < HH; ++k) acc += slast[k] * fc1_w[(size_t)p * HH + k];
    shh[p] = fmaxf(acc, 0.0f);
  }
  __syncthreads();

  if (tid < 2) {
    float acc = fc2_b[tid];
    for (int k = 0; k < HH; ++k) acc += shh[k] * fc2_w[(size_t)tid * HH + k];
    out[b * 2 + tid] = acc;
  }
}

// ---------------- host launch ----------------
extern "C" void kernel_launch(void* const* d_in, const int* in_sizes, int n_in,
                              void* d_out, int out_size, void* d_ws, size_t ws_size,
                              hipStream_t stream) {
  (void)in_sizes; (void)n_in; (void)out_size; (void)ws_size;
  const float* x     = (const float*)d_in[0];
  const float* W_ih0 = (const float*)d_in[1];
  const float* W_hh0 = (const float*)d_in[2];
  const float* b_ih0 = (const float*)d_in[3];
  const float* b_hh0 = (const float*)d_in[4];
  const float* W_ih1 = (const float*)d_in[5];
  const float* W_hh1 = (const float*)d_in[6];
  const float* b_ih1 = (const float*)d_in[7];
  const float* b_hh1 = (const float*)d_in[8];
  const float* se_w1 = (const float*)d_in[9];
  const float* se_b1 = (const float*)d_in[10];
  const float* se_w2 = (const float*)d_in[11];
  const float* se_b2 = (const float*)d_in[12];
  const float* fc1_w = (const float*)d_in[13];
  const float* fc1_b = (const float*)d_in[14];
  const float* fc2_w = (const float*)d_in[15];
  const float* fc2_b = (const float*)d_in[16];
  float* out = (float*)d_out;

  char* ws = (char*)d_ws;
  size_t off = 0;
  auto take = [&](size_t bytes) -> char* {
    char* p = ws + off;
    off = (off + bytes + 255) & ~(size_t)255;
    return p;
  };

  const long long NX    = (long long)BB * TT * KIN0;  // 16,384,000
  const long long NH1   = (long long)TT * BB * HH;    // 32,768,000
  const long long NWih0 = (long long)GG * KIN0;
  const long long NWhh  = (long long)GG * HH;

  unsigned short* x_hi    = (unsigned short*)take((size_t)NX * 2);
  unsigned short* x_lo    = (unsigned short*)take((size_t)NX * 2);
  unsigned short* wih0_hi = (unsigned short*)take((size_t)NWih0 * 2);
  unsigned short* wih0_lo = (unsigned short*)take((size_t)NWih0 * 2);
  unsigned short* whh0_hi = (unsigned short*)take((size_t)NWhh * 2);
  unsigned short* whh0_lo = (unsigned short*)take((size_t)NWhh * 2);
  unsigned short* wih1_hi = (unsigned short*)take((size_t)NWhh * 2);
  unsigned short* wih1_lo = (unsigned short*)take((size_t)NWhh * 2);
  unsigned short* whh1_hi = (unsigned short*)take((size_t)NWhh * 2);
  unsigned short* whh1_lo = (unsigned short*)take((size_t)NWhh * 2);
  unsigned short* h1_hi   = (unsigned short*)take((size_t)NH1 * 2);
  unsigned short* h1_lo   = (unsigned short*)take((size_t)NH1 * 2);
  float* osum  = (float*)take((size_t)BB * HH * 4);
  float* olast = (float*)take((size_t)BB * HH * 4);
  // zero-initialized block (contiguous; all sizes 256-aligned)
  char* zbase = ws + off;
  unsigned short* hb0_hi = (unsigned short*)take((size_t)2 * BB * HH * 2);
  unsigned short* hb0_lo = (unsigned short*)take((size_t)2 * BB * HH * 2);
  unsigned short* hb1_hi = (unsigned short*)take((size_t)2 * BB * HH * 2);
  unsigned short* hb1_lo = (unsigned short*)take((size_t)2 * BB * HH * 2);
  unsigned* bar          = (unsigned*)take(256);
  const long long zwords = (long long)((ws + off) - zbase) / 4;

  // 1) bf16 hi/lo plane decomposition of activations-in and all weights
  convert_planes<<<(unsigned)((NX + 255) / 256), 256, 0, stream>>>(x, x_hi, x_lo, NX);
  convert_planes<<<(unsigned)((NWih0 + 255) / 256), 256, 0, stream>>>(W_ih0, wih0_hi, wih0_lo, NWih0);
  convert_planes<<<(unsigned)((NWhh + 255) / 256), 256, 0, stream>>>(W_hh0, whh0_hi, whh0_lo, NWhh);
  convert_planes<<<(unsigned)((NWhh + 255) / 256), 256, 0, stream>>>(W_ih1, wih1_hi, wih1_lo, NWhh);
  convert_planes<<<(unsigned)((NWhh + 255) / 256), 256, 0, stream>>>(W_hh1, whh1_hi, whh1_lo, NWhh);
  zero_u32<<<(unsigned)((zwords + 255) / 256), 256, 0, stream>>>((unsigned*)zbase, zwords);

  // 2) layer 0 (persistent, all 1000 steps): x strides (b*T + t)*KIN0
  gru_kernel<KIN0, true><<<NWG, THREADS, 0, stream>>>(
      x_hi, x_lo, (long long)KIN0, (long long)TT * KIN0,
      wih0_hi, wih0_lo, whh0_hi, whh0_lo, b_ih0, b_hh0,
      hb0_hi, hb0_lo, h1_hi, h1_lo, nullptr, nullptr, bar);

  // 3) layer 1 (persistent): h1 strides (t*B + b)*HH; accumulates osum/olast
  gru_kernel<HH, false><<<NWG, THREADS, 0, stream>>>(
      h1_hi, h1_lo, (long long)BB * HH, (long long)HH,
      wih1_hi, wih1_lo, whh1_hi, whh1_lo, b_ih1, b_hh1,
      hb1_hi, hb1_lo, nullptr, nullptr, osum, olast, bar);

  // 4) SE + FC tail
  tail_kernel<<<BB, 256, 0, stream>>>(osum, olast, se_w1, se_b1, se_w2, se_b2,
                                      fc1_w, fc1_b, fc2_w, fc2_b, out);
}
